// AttentionDCA_55697135894764
// MI455X (gfx1250) — matrix operand order repaired and checked by reading, not compiled, not run on hardware
//
#include <hip/hip_runtime.h>
#include <hip/hip_bf16.h>
#include <math.h>

typedef __attribute__((ext_vector_type(16))) _Float16 v16h;
typedef __attribute__((ext_vector_type(8)))  _Float16 v8h;
typedef __attribute__((ext_vector_type(8)))  float    v8f;

#define Q_AA   21
#define D_REP  64
#define NH     32
#define LSEQ   256
#define DKDIM  32
#define DVDIM  32
#define MSEQ   8192
#define LAMBDA 1e-3f
#define KTOT   (LSEQ * Q_AA)   /* 5376 = 168*32, no K padding needed */
#define QPAD   32
#define RT     4               /* r-rows batched per block in k_gemm_E */

static __device__ __forceinline__ v8f wmma_f16(v16h a, v16h b, v8f c) {
  // D = A(16x32 f16) * B(32x16 f16) + C(16x16 f32)
  return __builtin_amdgcn_wmma_f32_16x16x32_f16(false, a, false, b, (short)0, c, false, false);
}

// ---------------------------------------------------------------- zero output
__global__ void k_zero(float* out) {
  if (threadIdx.x == 0 && blockIdx.x == 0) out[0] = 0.f;
}

// ---------------------------------------------------------------- Z -> u8
__global__ void k_pack_z8(const int* __restrict__ Z, unsigned char* __restrict__ Z8, int n) {
  int i = blockIdx.x * blockDim.x + threadIdx.x;
  if (i < n) Z8[i] = (unsigned char)Z[i];
}

// ---------------------------------------------------------------- attention P = softmax(QK^T/sqrt(dk))
__global__ void __launch_bounds__(128) k_attention(const float* __restrict__ Q,
                                                   const float* __restrict__ K,
                                                   float* __restrict__ P) {
  __shared__ float sS[16 * 256];
  __shared__ float sRed[16 * 8];
  __shared__ float sMax[16];
  __shared__ float sSum[16];
  const int h   = blockIdx.x >> 4;
  const int i0  = (blockIdx.x & 15) << 4;
  const int tid = threadIdx.x;
  const int lane = tid & 31;
  const int wid  = tid >> 5;
  const float scale = 0.17677669529663688f;  // 1/sqrt(32)

  // A-fragment from Q[h][i0..i0+15][0..31] (wave32 layout)
  const float* Qh = Q + ((size_t)h * LSEQ + i0) * DKDIM;
  v16h afrag;
  {
    int row = lane & 15;
    int d0  = (lane < 16) ? 0 : 16;
    const float* p = Qh + row * DKDIM + d0;
#pragma unroll
    for (int t = 0; t < 8; ++t) afrag[t] = (_Float16)p[t];
#pragma unroll
    for (int t = 0; t < 8; ++t) afrag[8 + t] = (_Float16)p[8 + t];
  }
  const float* Kh = K + (size_t)h * LSEQ * DKDIM;
  for (int ct = wid; ct < 16; ct += 4) {
    int j0 = ct << 4;
    v16h bfrag;  // B[d, j] = K[j, d]
    {
      int col = j0 + (lane & 15);
      int d0  = (lane < 16) ? 0 : 16;
      const float* p = Kh + col * DKDIM + d0;
#pragma unroll
      for (int t = 0; t < 16; ++t) bfrag[t] = (_Float16)p[t];
    }
    v8f c = {};
    c = wmma_f16(afrag, bfrag, c);
    int n     = lane & 15;
    int mbase = (lane < 16) ? 0 : 8;
#pragma unroll
    for (int v = 0; v < 8; ++v) sS[(mbase + v) * 256 + j0 + n] = c[v] * scale;
  }
  __syncthreads();
  // row softmax (16 rows x 256 cols), 8 segments per row
  int row = tid & 15, seg = tid >> 4;
  float pm = -1e30f;
  for (int c2 = seg * 32; c2 < seg * 32 + 32; ++c2) pm = fmaxf(pm, sS[row * 256 + c2]);
  sRed[row * 8 + seg] = pm;
  __syncthreads();
  if (tid < 16) { float m = -1e30f; for (int s = 0; s < 8; ++s) m = fmaxf(m, sRed[tid * 8 + s]); sMax[tid] = m; }
  __syncthreads();
  float rm = sMax[row];
  float ps = 0.f;
  for (int c2 = seg * 32; c2 < seg * 32 + 32; ++c2) ps += expf(sS[row * 256 + c2] - rm);
  sRed[row * 8 + seg] = ps;
  __syncthreads();
  if (tid < 16) { float s = 0.f; for (int t = 0; t < 8; ++t) s += sRed[tid * 8 + t]; sSum[tid] = s; }
  __syncthreads();
  float inv = 1.f / sSum[row];
  float* Pr = P + ((size_t)h * LSEQ + i0 + row) * LSEQ;
  for (int c2 = seg * 32; c2 < seg * 32 + 32; ++c2) Pr[c2] = expf(sS[row * 256 + c2] - rm) * inv;
}

// ---------------------------------------------------------------- A = 0.5*(P + P^T)
__global__ void k_sym(const float* __restrict__ P, float* __restrict__ A, int n) {
  int idx = blockIdx.x * blockDim.x + threadIdx.x;
  if (idx >= n) return;
  int j = idx & 255, i = (idx >> 8) & 255, h = idx >> 16;
  A[idx] = 0.5f * (P[idx] + P[((size_t)h * 256 + j) * 256 + i]);
}

// ---------------------------------------------------------------- Vaa RBF kernel per head
__global__ void __launch_bounds__(256) k_vaa(const float* __restrict__ reps,
                                             const float* __restrict__ Vm,
                                             float* __restrict__ Vaa) {
  __shared__ float sV1[Q_AA * DVDIM];
  __shared__ float sSq[Q_AA];
  int h = blockIdx.x, tid = threadIdx.x;
  const float* Vmh = Vm + (size_t)h * D_REP * DVDIM;
  for (int t = tid; t < Q_AA * DVDIM; t += 256) {
    int q = t / DVDIM, v = t % DVDIM;
    float acc = 0.f;
    for (int d = 0; d < D_REP; ++d) acc += reps[q * D_REP + d] * Vmh[d * DVDIM + v];
    sV1[t] = acc;
  }
  __syncthreads();
  if (tid < Q_AA) {
    float s = 0.f;
    for (int v = 0; v < DVDIM; ++v) { float x = sV1[tid * DVDIM + v]; s += x * x; }
    sSq[tid] = s;
  }
  __syncthreads();
  const float gamma = 1.f / (float)Q_AA;
  for (int t = tid; t < Q_AA * Q_AA; t += 256) {
    int q = t / Q_AA, a = t % Q_AA;
    float dot = 0.f;
    for (int v = 0; v < DVDIM; ++v) dot += sV1[q * DVDIM + v] * sV1[a * DVDIM + v];
    float d2 = sSq[q] + sSq[a] - 2.f * dot;
    Vaa[(size_t)h * Q_AA * Q_AA + t] = expf(-gamma * fmaxf(d2, 0.f));
  }
}

// ---------------------------------------------------------------- J16[r][q][j*21+a] (+ reg term)
__global__ void __launch_bounds__(256) k_build_J(const float* __restrict__ A,
                                                 const float* __restrict__ Vaa,
                                                 _Float16* __restrict__ J16,
                                                 float* __restrict__ out) {
  __shared__ float sVaa[NH * Q_AA * Q_AA];  // 56448 B
  __shared__ float sAh[NH];
  __shared__ float sRed[8];
  int r = blockIdx.x, tid = threadIdx.x;
  for (int t = tid; t < NH * Q_AA * Q_AA; t += 256) sVaa[t] = Vaa[t];
  _Float16* Jr = J16 + (size_t)r * QPAD * KTOT;
  for (int t = tid; t < (QPAD - Q_AA) * KTOT; t += 256)
    Jr[(size_t)Q_AA * KTOT + t] = (_Float16)0.f;  // zero q-pad rows
  __syncthreads();
  float regacc = 0.f;
  for (int j = 0; j < LSEQ; ++j) {
    if (tid < NH) sAh[tid] = A[((size_t)tid * LSEQ + r) * LSEQ + j];
    __syncthreads();
    for (int p = tid; p < Q_AA * Q_AA; p += 256) {
      float acc = 0.f;
#pragma unroll 8
      for (int h = 0; h < NH; ++h) acc += sAh[h] * sVaa[h * (Q_AA * Q_AA) + p];
      if (j == r) acc = 0.f;  // zero diagonal i==j
      int q = p / Q_AA, a = p - q * Q_AA;
      Jr[(size_t)q * KTOT + j * Q_AA + a] = (_Float16)acc;
      regacc += acc * acc;
    }
    __syncthreads();
  }
  for (int off = 16; off > 0; off >>= 1) regacc += __shfl_down(regacc, off, 32);
  if ((tid & 31) == 0) sRed[tid >> 5] = regacc;
  __syncthreads();
  if (tid == 0) {
    float s = 0.f;
    for (int w = 0; w < 8; ++w) s += sRed[w];
    atomicAdd(out, LAMBDA * s);
  }
}

// ---------------------------------------------------------------- E GEMM + logsumexp + PL accumulation
// Per block: RT r-rows, 128 m-columns. One one-hot B fragment feeds 2*RT WMMAs.
__global__ void __launch_bounds__(256) k_gemm_E(const _Float16* __restrict__ J16,
                                                const unsigned char* __restrict__ Z8,
                                                const float* __restrict__ wts,
                                                float* __restrict__ out) {
  __shared__ unsigned char sZ[LSEQ * 128];  // 32KB Z tile
  __shared__ float sRed[8];
  const int tid  = threadIdx.x, lane = tid & 31, wid = tid >> 5;
  const int r0   = (blockIdx.x >> 6) * RT;
  const int m0   = (blockIdx.x & 63) << 7;
  // stage Z[0..255][m0..m0+127] into LDS as bytes
  for (int d = tid; d < LSEQ * 32; d += 256) {
    int j = d >> 5, c = d & 31;
    ((unsigned int*)sZ)[j * 32 + c] = *(const unsigned int*)(Z8 + (size_t)j * MSEQ + m0 + c * 4);
  }
  __syncthreads();
  const int wcol = (wid << 4) + (lane & 15);  // column within the 128-wide tile
  const int arow = lane & 15;
  const int akb  = (lane < 16) ? 0 : 16;
  const int bkb  = (lane < 16) ? 0 : 16;
  const _Float16* Jbase = J16 + (size_t)r0 * QPAD * KTOT + (size_t)arow * KTOT + akb;
  v8f acc[RT][2];
#pragma unroll
  for (int rr = 0; rr < RT; ++rr) { acc[rr][0] = (v8f){}; acc[rr][1] = (v8f){}; }

  for (int k0 = 0; k0 < KTOT; k0 += 32) {
    // B fragment: synthesize one-hot column from Z bytes in LDS (built ONCE per k-step)
    v16h b;
    {
      unsigned int kk = (unsigned)(k0 + bkb);
      unsigned int j  = kk / 21u;
      unsigned int a  = kk - j * 21u;
#pragma unroll
      for (int t = 0; t < 16; ++t) {
        b[t] = (sZ[j * 128 + wcol] == (unsigned char)a) ? (_Float16)1.f : (_Float16)0.f;
        if (++a == 21u) { a = 0u; ++j; }
      }
    }
    if (k0 + 256 < KTOT) __builtin_prefetch(Jbase + k0 + 256, 0, 1);
    // feed B to 2*RT WMMAs (RT r-rows x 2 q-tiles); J streams from L2
#pragma unroll
    for (int rr = 0; rr < RT; ++rr) {
      const _Float16* p0 = Jbase + (size_t)rr * QPAD * KTOT + k0;
      v8h lo0 = *(const v8h*)p0;
      v8h hi0 = *(const v8h*)(p0 + 8);
      v16h a0 = __builtin_shufflevector(lo0, hi0, 0,1,2,3,4,5,6,7,8,9,10,11,12,13,14,15);
      const _Float16* p1 = p0 + (size_t)16 * KTOT;
      v8h lo1 = *(const v8h*)p1;
      v8h hi1 = *(const v8h*)(p1 + 8);
      v16h a1 = __builtin_shufflevector(lo1, hi1, 0,1,2,3,4,5,6,7,8,9,10,11,12,13,14,15);
      acc[rr][0] = wmma_f16(a0, b, acc[rr][0]);
      acc[rr][1] = wmma_f16(a1, b, acc[rr][1]);
    }
  }

  // Epilogue: per r-row, per column m: logsumexp over q=0..20 + correct-entry pick.
  // Lane<16 holds q=v (c0) and q=16+v (c1); lane>=16 holds q=8+v (c0).
  const int m = m0 + wcol;
  const float wm = wts[m];
  float total = 0.f;
#pragma unroll
  for (int rr = 0; rr < RT; ++rr) {
    v8f c0 = acc[rr][0], c1 = acc[rr][1];
    float lmax = -1e30f;
#pragma unroll
    for (int v = 0; v < 8; ++v) lmax = fmaxf(lmax, c0[v]);
    if (lane < 16) {
#pragma unroll
      for (int v = 0; v < 5; ++v) lmax = fmaxf(lmax, c1[v]);
    }
    float gmax = fmaxf(lmax, __shfl_xor(lmax, 16, 32));
    float lsum = 0.f;
#pragma unroll
    for (int v = 0; v < 8; ++v) lsum += expf(c0[v] - gmax);
    if (lane < 16) {
#pragma unroll
      for (int v = 0; v < 5; ++v) lsum += expf(c1[v] - gmax);
    }
    float lse = gmax + logf(lsum + __shfl_xor(lsum, 16, 32));
    int qc = (int)sZ[(r0 + rr) * 128 + wcol];  // Z[r, m]
    float own = -1e30f;
#pragma unroll
    for (int v = 0; v < 8; ++v) {
      int qheld = (lane < 16) ? v : (8 + v);
      own = (qheld == qc) ? c0[v] : own;
    }
#pragma unroll
    for (int v = 0; v < 5; ++v) {
      bool hit = (lane < 16) && ((16 + v) == qc);
      own = hit ? c1[v] : own;
    }
    float ec = fmaxf(own, __shfl_xor(own, 16, 32));
    total += (lane < 16) ? (wm * (ec - lse)) : 0.f;
  }
  for (int off = 16; off > 0; off >>= 1) total += __shfl_down(total, off, 32);
  if (lane == 0) sRed[wid] = total;
  __syncthreads();
  if (tid == 0) {
    float s = 0.f;
    for (int w = 0; w < 8; ++w) s += sRed[w];
    atomicAdd(out, -s);  // pl = -sum
  }
}

// ---------------------------------------------------------------- host launcher
extern "C" void kernel_launch(void* const* d_in, const int* in_sizes, int n_in,
                              void* d_out, int out_size, void* d_ws, size_t ws_size,
                              hipStream_t stream) {
  (void)in_sizes; (void)n_in; (void)out_size; (void)ws_size;
  const float* reps = (const float*)d_in[0];
  const float* Q    = (const float*)d_in[1];
  const float* K    = (const float*)d_in[2];
  const float* Vm   = (const float*)d_in[3];
  const int*   Z    = (const int*)d_in[4];
  const float* wts  = (const float*)d_in[5];
  float* out = (float*)d_out;

  char* ws = (char*)d_ws;
  float*         P   = (float*)(ws);                                          // 8 MB
  float*         A   = (float*)(ws + ((size_t)8 << 20));                      // 8 MB
  float*         Vaa = (float*)(ws + ((size_t)16 << 20));                     // 56 KB
  unsigned char* Z8  = (unsigned char*)(ws + ((size_t)16 << 20) + (1u << 16)); // 2 MB
  _Float16*      J16 = (_Float16*)(ws + ((size_t)16 << 20) + (1u << 16) + ((size_t)2 << 20)); // 88 MB

  const int NZ = LSEQ * MSEQ;        // 2,097,152
  const int NA = NH * LSEQ * LSEQ;   // 2,097,152

  k_zero<<<1, 1, 0, stream>>>(out);
  k_pack_z8<<<(NZ + 255) / 256, 256, 0, stream>>>(Z, Z8, NZ);
  k_vaa<<<NH, 256, 0, stream>>>(reps, Vm, Vaa);
  k_attention<<<NH * 16, 128, 0, stream>>>(Q, K, P);
  k_sym<<<(NA + 255) / 256, 256, 0, stream>>>(P, A, NA);
  k_build_J<<<LSEQ, 256, 0, stream>>>(A, Vaa, J16, out);
  k_gemm_E<<<(LSEQ / RT) * 64, 256, 0, stream>>>(J16, Z8, wts, out);
}
